// CA_Block_V3_DAV_55190329754126
// MI455X (gfx1250) — matrix operand order repaired and checked by reading, not compiled, not run on hardware
//
#include <hip/hip_runtime.h>
#include <hip/hip_bf16.h>
#include <math.h>

typedef __attribute__((ext_vector_type(16))) _Float16 v16h;
typedef __attribute__((ext_vector_type(8)))  _Float16 v8h;
typedef __attribute__((ext_vector_type(8)))  float    v8f;

#define Bq   2
#define Cc   128
#define Hh   96
#define Ww   320
#define CQ   32
#define Dd   32
#define HW   (Hh * Ww)          // 30720
#define NPix (Bq * HW)          // 61440
#define NTILES (NPix / 16)      // 3840 (exact)
#define WAVES_PER_BLOCK 8

// Per-32-K-block storage order so one contiguous v16h load yields the
// A fragment of v_wmma_f32_16x16x32_f16 (contiguous k order, matching B):
//   stored[kt*32 + 16*half + e] == W[kt*32 + 16*half + e]
__device__ __forceinline__ int perm32(int j) {
    return j;
}

__device__ __forceinline__ int reflect_idx(int v, int L) {
    v = (v < 0) ? -v : v;
    return (v >= L) ? (2 * L - 2 - v) : v;
}

// ---------------------------------------------------------------------------
// 1) Pack f32 NCHW feature -> f16 pixel-major [n][C], 16B vector stores
// ---------------------------------------------------------------------------
__global__ void pack_feat_kernel(const float* __restrict__ x, _Float16* __restrict__ o) {
    int idx = blockIdx.x * blockDim.x + threadIdx.x;     // over Bq*HW*(Cc/8)
    if (idx >= NPix * (Cc / 8)) return;
    int p  = idx % HW;
    int cg = (idx / HW) % (Cc / 8);
    int b  = idx / (HW * (Cc / 8));
    const float* xp = x + (size_t)b * Cc * HW + (size_t)(cg * 8) * HW + p;
    v8h vv;
    #pragma unroll
    for (int e = 0; e < 8; ++e) vv[e] = (_Float16)xp[(size_t)e * HW];
    *(v8h*)(o + (size_t)(b * HW + p) * Cc + cg * 8) = vv;
}

// ---------------------------------------------------------------------------
// 2) Pack GEMM weights f32 [M][Ktot] -> f16 fragment-ordered [M][Ktot]
// ---------------------------------------------------------------------------
__global__ void pack_w_kernel(const float* __restrict__ w, _Float16* __restrict__ o,
                              int M, int Ktot) {
    int idx = blockIdx.x * blockDim.x + threadIdx.x;
    if (idx >= M * Ktot) return;
    int m  = idx / Ktot;
    int kk = idx % Ktot;
    int kt = kk / 32, j = kk % 32;
    o[idx] = (_Float16)w[m * Ktot + kt * 32 + perm32(j)];
}

// rc_w (O,C,3,3) -> f16 [tap][O][C] fragment-ordered in C
__global__ void pack_rcw_kernel(const float* __restrict__ w, _Float16* __restrict__ o) {
    int idx = blockIdx.x * blockDim.x + threadIdx.x;     // over 9*128*128
    if (idx >= 9 * Cc * Cc) return;
    int tap = idx / (Cc * Cc);
    int rem = idx % (Cc * Cc);
    int m   = rem / Cc;
    int kk  = rem % Cc;
    int kt  = kk / 32, j = kk % 32;
    int cix = kt * 32 + perm32(j);
    o[idx] = (_Float16)w[(((size_t)m * Cc + cix) * 3 + tap / 3) * 3 + tap % 3];
}

// ---------------------------------------------------------------------------
// 3) q/k GEMM: out[32 x N] = W[32x128] * Xp^T, WMMA, packed-f16 A fragments
// ---------------------------------------------------------------------------
__global__ void gemm_qk_kernel(const _Float16* __restrict__ Xp,
                               const _Float16* __restrict__ Wp,  // packed 32x128
                               const float* __restrict__ bg,
                               float* __restrict__ out) {        // [B][32][HW]
    int wave = (blockIdx.x * blockDim.x + threadIdx.x) >> 5;
    int lane = threadIdx.x & 31;
    if (wave >= NTILES) return;
    int n    = wave * 16 + (lane & 15);
    int half = lane >> 4;
    int mrow = lane & 15;

    v8f c0 = {}, c1 = {};
    const _Float16* brow = Xp + (size_t)n * Cc;
    #pragma unroll
    for (int kt = 0; kt < 4; ++kt) {
        int kb = kt * 32 + 16 * half;
        v16h a0 = *(const v16h*)(Wp + (size_t)mrow * Cc + kb);
        v16h a1 = *(const v16h*)(Wp + (size_t)(mrow + 16) * Cc + kb);
        v16h bf = *(const v16h*)(brow + kb);
        c0 = __builtin_amdgcn_wmma_f32_16x16x32_f16(false, a0, false, bf, (short)0, c0, false, false);
        c1 = __builtin_amdgcn_wmma_f32_16x16x32_f16(false, a1, false, bf, (short)0, c1, false, false);
    }
    int b = n / HW, p = n % HW;
    float* ob = out + (size_t)b * CQ * HW + p;
    #pragma unroll
    for (int r = 0; r < 8; ++r) {
        int m0 = r + 8 * half;
        ob[(size_t)m0 * HW]        = c0[r] + bg[m0];
        ob[(size_t)(m0 + 16) * HW] = c1[r] + bg[m0 + 16];
    }
}

// ---------------------------------------------------------------------------
// 4) per-pixel q norm
// ---------------------------------------------------------------------------
__global__ void qnorm_kernel(const float* __restrict__ q, float* __restrict__ qn) {
    int n = blockIdx.x * blockDim.x + threadIdx.x;
    if (n >= NPix) return;
    int b = n / HW, p = n % HW;
    const float* qp = q + (size_t)b * CQ * HW + p;
    float s = 0.f;
    #pragma unroll
    for (int c = 0; c < CQ; ++c) { float v = qp[(size_t)c * HW]; s += v * v; }
    qn[n] = fmaxf(sqrtf(s), 1e-12f);
}

// ---------------------------------------------------------------------------
// 5) warp + normalized correlation -> sim[b][d][h][w]
// ---------------------------------------------------------------------------
__global__ void sim_kernel(const float* __restrict__ q, const float* __restrict__ k,
                           const float* __restrict__ qn, const float* __restrict__ disp,
                           const float* __restrict__ directs, float* __restrict__ sim) {
    int idx = blockIdx.x * blockDim.x + threadIdx.x;     // Bq*Dd*HW
    if (idx >= Bq * Dd * HW) return;
    int w = idx % Ww;
    int h = (idx / Ww) % Hh;
    int d = (idx / HW) % Dd;
    int b = idx / (HW * Dd);

    float pos = (float)w + disp[d] * directs[b] * (float)(Ww - 1);
    pos = fminf(fmaxf(pos, 0.f), (float)(Ww - 1));
    int x0 = (int)floorf(pos);
    x0 = min(max(x0, 0), Ww - 1);
    int x1 = min(x0 + 1, Ww - 1);
    float t = pos - (float)x0;

    const float* kb = k + (size_t)b * CQ * HW + h * Ww;
    const float* qp = q + (size_t)b * CQ * HW + h * Ww + w;
    float dot = 0.f, ss = 0.f;
    #pragma unroll
    for (int c = 0; c < CQ; ++c) {
        size_t co = (size_t)c * HW;
        float g = kb[co + x0] * (1.f - t) + kb[co + x1] * t;
        dot += qp[co] * g;
        ss  += g * g;
    }
    float knorm = fmaxf(sqrtf(ss), 1e-12f);
    sim[idx] = dot / (qn[b * HW + h * Ww + w] * knorm);
}

// ---------------------------------------------------------------------------
// 6) 3x3x3 cost filter (zero pad)
// ---------------------------------------------------------------------------
__global__ void cost_kernel(const float* __restrict__ sim, const float* __restrict__ cfw,
                            const float* __restrict__ cfb, float* __restrict__ cost) {
    int idx = blockIdx.x * blockDim.x + threadIdx.x;
    if (idx >= Bq * Dd * HW) return;
    int w = idx % Ww;
    int h = (idx / Ww) % Hh;
    int d = (idx / HW) % Dd;
    int b = idx / (HW * Dd);
    float acc = cfb[0];
    #pragma unroll
    for (int dz = -1; dz <= 1; ++dz)
        #pragma unroll
        for (int dy = -1; dy <= 1; ++dy)
            #pragma unroll
            for (int dx = -1; dx <= 1; ++dx) {
                int dd = d + dz, hh = h + dy, ww = w + dx;
                if (dd >= 0 && dd < Dd && hh >= 0 && hh < Hh && ww >= 0 && ww < Ww) {
                    acc += cfw[(dz + 1) * 9 + (dy + 1) * 3 + (dx + 1)] *
                           sim[(((size_t)b * Dd + dd) * Hh + hh) * Ww + ww];
                }
            }
    cost[idx] = acc;
}

// ---------------------------------------------------------------------------
// 7) softmax over D -> norm_cost (second output)
// ---------------------------------------------------------------------------
__global__ void softmax_kernel(const float* __restrict__ cost, float* __restrict__ nc) {
    int n = blockIdx.x * blockDim.x + threadIdx.x;
    if (n >= NPix) return;
    int b = n / HW, p = n % HW;
    const float* cp = cost + (size_t)b * Dd * HW + p;
    float m = -1e30f;
    #pragma unroll
    for (int d = 0; d < Dd; ++d) m = fmaxf(m, cp[(size_t)d * HW]);
    float s = 0.f;
    #pragma unroll
    for (int d = 0; d < Dd; ++d) s += expf(cp[(size_t)d * HW] - m);
    float inv = 1.f / s;
    float* op = nc + (size_t)b * Dd * HW + p;
    #pragma unroll
    for (int d = 0; d < Dd; ++d) op[(size_t)d * HW] = expf(cp[(size_t)d * HW] - m) * inv;
}

// ---------------------------------------------------------------------------
// 8) batch-wide sum / sum-of-squares (groupnorm stats)
// ---------------------------------------------------------------------------
__global__ void zero_kernel(float* __restrict__ a, int n) {
    int i = blockIdx.x * blockDim.x + threadIdx.x;
    if (i < n) a[i] = 0.f;
}

__global__ void stats_kernel(const float* __restrict__ x, float* __restrict__ acc, int L) {
    int b = blockIdx.y;
    const float* xb = x + (size_t)b * L;
    float s = 0.f, ss = 0.f;
    for (int i = blockIdx.x * blockDim.x + threadIdx.x; i < L; i += gridDim.x * blockDim.x) {
        float v = xb[i]; s += v; ss += v * v;
    }
    __shared__ float sh0[256];
    __shared__ float sh1[256];
    sh0[threadIdx.x] = s; sh1[threadIdx.x] = ss;
    __syncthreads();
    for (int off = 128; off > 0; off >>= 1) {
        if (threadIdx.x < off) {
            sh0[threadIdx.x] += sh0[threadIdx.x + off];
            sh1[threadIdx.x] += sh1[threadIdx.x + off];
        }
        __syncthreads();
    }
    if (threadIdx.x == 0) {
        atomicAdd(&acc[b * 2 + 0], sh0[0]);
        atomicAdd(&acc[b * 2 + 1], sh1[0]);
    }
}

// ---------------------------------------------------------------------------
// 9) groupnorm(norm_cost) -> f16 packed [n][32]
// ---------------------------------------------------------------------------
__global__ void pack_cn_kernel(const float* __restrict__ nc, const float* __restrict__ acc,
                               const float* __restrict__ ncw, const float* __restrict__ ncb,
                               _Float16* __restrict__ cpk) {
    int idx = blockIdx.x * blockDim.x + threadIdx.x;
    if (idx >= Bq * Dd * HW) return;
    int p = idx % HW;
    int d = (idx / HW) % Dd;
    int b = idx / (HW * Dd);
    float cnt = (float)(Dd * HW);
    float mu  = acc[b * 2] / cnt;
    float var = acc[b * 2 + 1] / cnt - mu * mu;
    float inv = rsqrtf(var + 1e-5f);
    cpk[(size_t)(b * HW + p) * Dd + d] = (_Float16)((nc[idx] - mu) * inv * ncw[d] + ncb[d]);
}

// ---------------------------------------------------------------------------
// 10) cost-embedding GEMM: [128x32] x [32 x N] via WMMA, K=32 single step
// ---------------------------------------------------------------------------
__global__ void gemm_ce_kernel(const _Float16* __restrict__ cpk, const _Float16* __restrict__ Wp,
                               const float* __restrict__ bce, float* __restrict__ out) {
    int wave = (blockIdx.x * blockDim.x + threadIdx.x) >> 5;
    int lane = threadIdx.x & 31;
    if (wave >= NTILES) return;
    int mp   = blockIdx.y;                 // 0..3 -> rows [mp*32, mp*32+32)
    int n    = wave * 16 + (lane & 15);
    int half = lane >> 4;
    int mrow = lane & 15;

    int kb = 16 * half;
    v16h a0 = *(const v16h*)(Wp + (size_t)(mp * 32 + mrow) * Dd + kb);
    v16h a1 = *(const v16h*)(Wp + (size_t)(mp * 32 + 16 + mrow) * Dd + kb);
    v16h bf = *(const v16h*)(cpk + (size_t)n * Dd + kb);
    v8f c0 = {}, c1 = {};
    c0 = __builtin_amdgcn_wmma_f32_16x16x32_f16(false, a0, false, bf, (short)0, c0, false, false);
    c1 = __builtin_amdgcn_wmma_f32_16x16x32_f16(false, a1, false, bf, (short)0, c1, false, false);

    int b = n / HW, p = n % HW;
    float* ob = out + (size_t)b * Cc * HW + p;
    #pragma unroll
    for (int r = 0; r < 8; ++r) {
        int m0 = mp * 32 + r + 8 * half;
        ob[(size_t)m0 * HW]        = c0[r] + bce[m0];
        ob[(size_t)(m0 + 16) * HW] = c1[r] + bce[m0 + 16];
    }
}

// ---------------------------------------------------------------------------
// 11) groupnorm(t_feat) + gated fusion -> f16 packed [n][128], 32B stores
// ---------------------------------------------------------------------------
__global__ void fusion_kernel(const float* __restrict__ tfeat, const float* __restrict__ accT,
                              const float* __restrict__ ntw, const float* __restrict__ ntb,
                              const float* __restrict__ cfeat, const float* __restrict__ faw,
                              const float* __restrict__ fab, _Float16* __restrict__ fpk) {
    int n = blockIdx.x * blockDim.x + threadIdx.x;
    if (n >= NPix) return;
    int b = n / HW, p = n % HW;
    float cnt = (float)(Cc * HW);
    float mu  = accT[b * 2] / cnt;
    float var = accT[b * 2 + 1] / cnt - mu * mu;
    float inv = rsqrtf(var + 1e-5f);

    const float* tp = tfeat + (size_t)b * Cc * HW + p;
    const float* cp = cfeat + (size_t)b * Cc * HW + p;
    float acc = fab[0];
    for (int c = 0; c < Cc; ++c) {
        float tn = (tp[(size_t)c * HW] - mu) * inv * ntw[c] + ntb[c];
        acc += faw[c] * tn + faw[Cc + c] * cp[(size_t)c * HW];
    }
    float alpha = 1.f / (1.f + expf(-acc));
    _Float16* op = fpk + (size_t)n * Cc;
    for (int cg = 0; cg < Cc; cg += 16) {
        v16h vv;
        #pragma unroll
        for (int e = 0; e < 16; ++e) {
            int c = cg + e;
            float tn = (tp[(size_t)c * HW] - mu) * inv * ntw[c] + ntb[c];
            vv[e] = (_Float16)(alpha * tn + (1.f - alpha) * cp[(size_t)c * HW]);
        }
        *(v16h*)(op + cg) = vv;
    }
}

// ---------------------------------------------------------------------------
// 12) 3x3 conv 128->128 as implicit GEMM (reflect pad), WMMA
//     A fragments staged in LDS via GLOBAL_LOAD_ASYNC_TO_LDS_B128 (ASYNCcnt);
//     B rows prefetched one tap ahead (global_prefetch_b8)
// ---------------------------------------------------------------------------
__global__ void conv_rc_kernel(const _Float16* __restrict__ fpk, const _Float16* __restrict__ wrcp,
                               const float* __restrict__ brc, float* __restrict__ xout) {
    __shared__ __align__(32) _Float16 wsh[32 * Cc];      // 8 KB: current tap, 32 rows
    int wave = (blockIdx.x * blockDim.x + threadIdx.x) >> 5;
    int lane = threadIdx.x & 31;
    int mp   = blockIdx.y;                               // 0..3
    int n    = wave * 16 + (lane & 15);                  // grid sized exactly
    int half = lane >> 4;
    int mrow = lane & 15;
    int b = n / HW, p = n % HW, h = p / Ww, w = p % Ww;

    unsigned lds_base = (unsigned)(uintptr_t)(&wsh[0]);  // LDS byte address

    v8f c0 = {}, c1 = {};
    for (int tap = 0; tap < 9; ++tap) {
        int ky = tap / 3, kx = tap % 3;
        __syncthreads();
        {   // async DMA memory -> LDS: 512 x 16B chunks, each wave issues 2
            const char* src = (const char*)(wrcp + ((size_t)tap * Cc + mp * 32) * Cc);
            for (int i = threadIdx.x; i < (32 * Cc * 2) / 16; i += blockDim.x) {
                unsigned ldsa = lds_base + (unsigned)i * 16u;
                unsigned long long ga = (unsigned long long)(uintptr_t)(src + (size_t)i * 16);
                asm volatile("global_load_async_to_lds_b128 %0, %1, off"
                             :: "v"(ldsa), "v"(ga) : "memory");
            }
            asm volatile("s_wait_asynccnt 0x0" ::: "memory");
        }
        __syncthreads();

        int hh = reflect_idx(h + ky - 1, Hh);
        int ww = reflect_idx(w + kx - 1, Ww);
        const _Float16* brow = fpk + (size_t)(b * HW + hh * Ww + ww) * Cc;

        if (tap < 8) {  // prefetch next tap's B row (global_prefetch_b8)
            int kyn = (tap + 1) / 3, kxn = (tap + 1) % 3;
            int hn = reflect_idx(h + kyn - 1, Hh);
            int wn = reflect_idx(w + kxn - 1, Ww);
            __builtin_prefetch(fpk + (size_t)(b * HW + hn * Ww + wn) * Cc, 0, 1);
        }

        #pragma unroll
        for (int kt = 0; kt < 4; ++kt) {
            int kb = kt * 32 + 16 * half;
            v16h a0 = *(const v16h*)(wsh + (size_t)mrow * Cc + kb);
            v16h a1 = *(const v16h*)(wsh + (size_t)(mrow + 16) * Cc + kb);
            v16h bf = *(const v16h*)(brow + kb);
            c0 = __builtin_amdgcn_wmma_f32_16x16x32_f16(false, a0, false, bf, (short)0, c0, false, false);
            c1 = __builtin_amdgcn_wmma_f32_16x16x32_f16(false, a1, false, bf, (short)0, c1, false, false);
        }
    }
    float* ob = xout + (size_t)b * Cc * HW + p;
    #pragma unroll
    for (int r = 0; r < 8; ++r) {
        int m0 = mp * 32 + r + 8 * half;
        ob[(size_t)m0 * HW]        = c0[r] + brc[m0];
        ob[(size_t)(m0 + 16) * HW] = c1[r] + brc[m0 + 16];
    }
}

// ---------------------------------------------------------------------------
// 13) SE: spatial mean per (b, o)
// ---------------------------------------------------------------------------
__global__ void semean_kernel(const float* __restrict__ x, float* __restrict__ ym) {
    int bo = blockIdx.x;
    const float* xp = x + (size_t)bo * HW;
    float s = 0.f;
    for (int i = threadIdx.x; i < HW; i += blockDim.x) s += xp[i];
    __shared__ float sh[256];
    sh[threadIdx.x] = s;
    __syncthreads();
    for (int off = 128; off > 0; off >>= 1) {
        if (threadIdx.x < off) sh[threadIdx.x] += sh[threadIdx.x + off];
        __syncthreads();
    }
    if (threadIdx.x == 0) ym[bo] = sh[0] / (float)HW;
}

// ---------------------------------------------------------------------------
// 14) SE MLP: relu(128->8) then sigmoid(8->128)
// ---------------------------------------------------------------------------
__global__ void semlp_kernel(const float* __restrict__ ym, const float* __restrict__ w1,
                             const float* __restrict__ w2, float* __restrict__ yse) {
    __shared__ float hsh[Bq * 8];
    int t = threadIdx.x;
    if (t < Bq * 8) {
        int b = t / 8, j = t % 8;
        float s = 0.f;
        for (int c = 0; c < Cc; ++c) s += w1[j * Cc + c] * ym[b * Cc + c];
        hsh[t] = fmaxf(s, 0.f);
    }
    __syncthreads();
    if (t < Bq * Cc) {
        int b = t / Cc, o = t % Cc;
        float s = 0.f;
        #pragma unroll
        for (int j = 0; j < 8; ++j) s += w2[o * 8 + j] * hsh[b * 8 + j];
        yse[t] = 1.f / (1.f + expf(-s));
    }
}

// ---------------------------------------------------------------------------
// 15) final: elu(x * y)
// ---------------------------------------------------------------------------
__global__ void final_kernel(const float* __restrict__ x, const float* __restrict__ yse,
                             float* __restrict__ out) {
    int idx = blockIdx.x * blockDim.x + threadIdx.x;
    if (idx >= Bq * Cc * HW) return;
    int bo = idx / HW;
    float v = x[idx] * yse[bo];
    out[idx] = (v > 0.f) ? v : expm1f(v);
}

// ---------------------------------------------------------------------------
extern "C" void kernel_launch(void* const* d_in, const int* in_sizes, int n_in,
                              void* d_out, int out_size, void* d_ws, size_t ws_size,
                              hipStream_t stream) {
    const float* t_feat  = (const float*)d_in[0];
    const float* s_feat  = (const float*)d_in[1];
    const float* directs = (const float*)d_in[2];
    const float* disp    = (const float*)d_in[3];
    const float* q_w = (const float*)d_in[4];  const float* q_b = (const float*)d_in[5];
    const float* k_w = (const float*)d_in[6];  const float* k_b = (const float*)d_in[7];
    const float* cf_w = (const float*)d_in[8]; const float* cf_b = (const float*)d_in[9];
    const float* nt_w = (const float*)d_in[10]; const float* nt_b = (const float*)d_in[11];
    const float* nc_w = (const float*)d_in[12]; const float* nc_b = (const float*)d_in[13];
    const float* ce_w = (const float*)d_in[14]; const float* ce_b = (const float*)d_in[15];
    const float* fa_w = (const float*)d_in[16]; const float* fa_b = (const float*)d_in[17];
    const float* rc_w = (const float*)d_in[18]; const float* rc_b = (const float*)d_in[19];
    const float* se_w1 = (const float*)d_in[20]; const float* se_w2 = (const float*)d_in[21];

    float* out_x  = (float*)d_out;
    float* out_nc = out_x + (size_t)Bq * Cc * HW;

    char* ws = (char*)d_ws;
    _Float16* tpk  = (_Float16*)(ws + 0);                     // 15,728,640 B
    _Float16* spk  = (_Float16*)(ws + 15728640);              // 15,728,640 B (reused as fpk)
    float*    qbuf = (float*)   (ws + 31457280);              //  7,864,320 B (reused as cost)
    float*    kbuf = (float*)   (ws + 39321600);              //  7,864,320 B (reused as cpk)
    float*    qn   = (float*)   (ws + 47185920);              //    245,760 B
    float*    simb = (float*)   (ws + 47431680);              //  7,864,320 B
    float*    cfeat= (float*)   (ws + 55296000);              // 31,457,280 B
    float*    xbuf = (float*)   (ws + 86753280);              // 31,457,280 B
    float*    stats= (float*)   (ws + 118210560);             // 8 floats (+pad)
    float*    ym   = (float*)   (ws + 118210816);
    float*    yse  = (float*)   (ws + 118211840);
    _Float16* wq_p = (_Float16*)(ws + 118212864);             // 8,192 B
    _Float16* wk_p = (_Float16*)(ws + 118221056);             // 8,192 B
    _Float16* wce_p= (_Float16*)(ws + 118229248);             // 8,192 B
    _Float16* wrc_p= (_Float16*)(ws + 118237440);             // 294,912 B
    float*    accT = stats;
    float*    accC = stats + 4;
    float*    costb = qbuf;             // reuse after sim
    _Float16* cpk   = (_Float16*)kbuf;  // reuse after sim
    _Float16* fpk   = spk;              // reuse after q/k GEMMs

    const int elems_full = Bq * Cc * HW;   // 7,864,320
    const int elems_vol  = Bq * Dd * HW;   // 1,966,080

    // one-time packs: features -> f16 [n][C]; weights -> f16 fragment order
    pack_feat_kernel<<<(NPix * (Cc / 8) + 255) / 256, 256, 0, stream>>>(t_feat, tpk);
    pack_feat_kernel<<<(NPix * (Cc / 8) + 255) / 256, 256, 0, stream>>>(s_feat, spk);
    pack_w_kernel<<<(CQ * Cc + 255) / 256, 256, 0, stream>>>(q_w, wq_p, CQ, Cc);
    pack_w_kernel<<<(CQ * Cc + 255) / 256, 256, 0, stream>>>(k_w, wk_p, CQ, Cc);
    pack_w_kernel<<<(Cc * Dd + 255) / 256, 256, 0, stream>>>(ce_w, wce_p, Cc, Dd);
    pack_rcw_kernel<<<(9 * Cc * Cc + 255) / 256, 256, 0, stream>>>(rc_w, wrc_p);

    // q/k 1x1 GEMMs (WMMA)
    int gemm_blocks = NTILES / WAVES_PER_BLOCK;   // 480
    gemm_qk_kernel<<<gemm_blocks, 32 * WAVES_PER_BLOCK, 0, stream>>>(tpk, wq_p, q_b, qbuf);
    gemm_qk_kernel<<<gemm_blocks, 32 * WAVES_PER_BLOCK, 0, stream>>>(spk, wk_p, k_b, kbuf);

    // normalized warped correlation
    qnorm_kernel<<<(NPix + 255) / 256, 256, 0, stream>>>(qbuf, qn);
    sim_kernel<<<(elems_vol + 255) / 256, 256, 0, stream>>>(qbuf, kbuf, qn, disp, directs, simb);

    // cost filtering + softmax (second output)
    cost_kernel<<<(elems_vol + 255) / 256, 256, 0, stream>>>(simb, cf_w, cf_b, costb);
    softmax_kernel<<<(NPix + 255) / 256, 256, 0, stream>>>(costb, out_nc);

    // groupnorm statistics
    zero_kernel<<<1, 64, 0, stream>>>(stats, 8);
    {
        dim3 g(512, Bq);
        stats_kernel<<<g, 256, 0, stream>>>(t_feat, accT, Cc * HW);
        stats_kernel<<<g, 256, 0, stream>>>(out_nc, accC, Dd * HW);
    }

    // c_n -> f16 packed, then cost-embedding GEMM (WMMA)
    pack_cn_kernel<<<(elems_vol + 255) / 256, 256, 0, stream>>>(out_nc, accC, nc_w, nc_b, cpk);
    {
        dim3 g(gemm_blocks, 4);
        gemm_ce_kernel<<<g, 32 * WAVES_PER_BLOCK, 0, stream>>>(cpk, wce_p, ce_b, cfeat);
    }

    // t_n + gated fusion -> f16 packed [n][128]
    fusion_kernel<<<(NPix + 255) / 256, 256, 0, stream>>>(t_feat, accT, nt_w, nt_b,
                                                          cfeat, fa_w, fa_b, fpk);

    // 3x3 conv as implicit GEMM (WMMA + async-LDS weight staging + prefetch)
    {
        dim3 g(gemm_blocks, 4);
        conv_rc_kernel<<<g, 32 * WAVES_PER_BLOCK, 0, stream>>>(fpk, wrc_p, rc_b, xbuf);
    }

    // squeeze-excite + ELU -> first output
    semean_kernel<<<Bq * Cc, 256, 0, stream>>>(xbuf, ym);
    semlp_kernel<<<1, 256, 0, stream>>>(ym, se_w1, se_w2, yse);
    final_kernel<<<(elems_full + 255) / 256, 256, 0, stream>>>(xbuf, yse, out_x);
}
